// DNDLSTM_54056458387478
// MI455X (gfx1250) — matrix operand order repaired and checked by reading, compiled, not run on hardware
//
#include <hip/hip_runtime.h>
#include <hip/hip_bf16.h>
#include <math.h>

// ---------------------------------------------------------------------------
// Types for CDNA5 WMMA
// ---------------------------------------------------------------------------
typedef __bf16 bf16;
typedef bf16  v16bf __attribute__((ext_vector_type(16)));
typedef float v8f   __attribute__((ext_vector_type(8)));

// Problem constants
#define HSZ      1024
#define DIN      2048
#define KEY_DIM  1024
#define DICT_LEN 16384
#define A2C_H    512
#define N_ACT    1024
#define BATCH    256
#define EPSV     1e-8f

// Output layout (flat floats, reference return order)
#define O_A     0            // a_t              (256)
#define O_PROB  256          // prob_a_t         (256)
#define O_V     512          // v_t              (256)
#define O_ENT   768          // entropy          (256)
#define O_H     1024         // h_t              (256*1024)
#define O_C     263168       // c_t              (256*1024)
#define O_BEST  525312       // best_mem_id      (256)
#define O_NEWK  525568       // new_keys         (16384*1024)
#define O_NEWV  17302784     // new_vals         (16384*1024)

// Workspace layout (floats)
#define W_PREACT 0                        // 256*5120
#define W_SIMS   1310720                  // 256*16384
#define W_QN     5505024                  // 256*1024
#define W_KINV   5767168                  // 16384
#define W_HA     5783552                  // 256*512
#define W_LOGITS 5914624                  // 256*1024
#define W_BEST   6176768                  // 256 ints

// ---------------------------------------------------------------------------
// WMMA fragment loaders (bf16, 16x16x32). LDS rows are 16B aligned (stride 40
// halves = 80B).  A-matrix 16x32 layout: lane l (h=l>>4, m=l&15):
//   elems 0..7  -> K = h*8 + (0..7)
//   elems 8..15 -> K = 16 + h*8 + (0..7)
// B-matrix 32x16 (stored transposed in LDS as [n][k]): lane l (n=l&15):
//   elems 0..7  -> K = h*8 + (0..7)
//   elems 8..15 -> K = 16 + h*8 + (0..7)
// ---------------------------------------------------------------------------
__device__ inline v16bf load_fragA(const bf16* row, int h) {
  union { v16bf v; uint4 q[2]; } r;
  r.q[0] = *(const uint4*)(row + h * 8);
  r.q[1] = *(const uint4*)(row + 16 + h * 8);
  return r.v;
}
__device__ inline v16bf load_fragB(const bf16* row, int h) {
  union { v16bf v; uint4 q[2]; } r;
  r.q[0] = *(const uint4*)(row + h * 8);
  r.q[1] = *(const uint4*)(row + 16 + h * 8);
  return r.v;
}

// ---------------------------------------------------------------------------
// Generic bf16-WMMA GEMM:  C[M][N] = act( A @ B + bias1 + bias2 ) * nscale
// A has up to two row-major segments A1[M][K1], A2[M][K2] (K concatenated).
// B: if BTRANS, B1 is [N][K1] row-major (we compute A @ B1^T);
//    else       B1/B2 are [K][N] row-major.
// Block tile 64x64, K-step 32, 256 threads = 8 waves, EXEC uniform.
// ---------------------------------------------------------------------------
template<bool BTRANS, bool RELU, bool NSCALE>
__global__ __launch_bounds__(256)
void gemm_wmma_bf16(const float* __restrict__ A1, const float* __restrict__ A2,
                    int K1, int K2,
                    const float* __restrict__ B1, const float* __restrict__ B2,
                    const float* __restrict__ bias1, const float* __restrict__ bias2,
                    const float* __restrict__ nscale,
                    float* __restrict__ C, int N)
{
  __shared__ __align__(16) bf16 lsA[64][40];
  __shared__ __align__(16) bf16 lsB[64][40];   // transposed: [n][k]

  const int tid = threadIdx.x;
  const int w   = tid >> 5;
  const int l   = tid & 31;
  const int h   = l >> 4;
  const int ln  = l & 15;
  const int wm  = w & 3;            // wave's 16-row stripe within 64
  const int wnb = (w >> 2) << 1;    // wave's first 16-col tile (0 or 2)
  const int bm  = blockIdx.y * 64;
  const int bn  = blockIdx.x * 64;
  const int K   = K1 + K2;

  const int arow = tid >> 2;              // 0..63
  const int acg  = (tid & 3) << 3;        // 0,8,16,24
  const int bkr  = tid >> 3;              // 0..31
  const int bng  = (tid & 7) << 3;        // 0..56

  v8f acc0 = {0.f,0.f,0.f,0.f,0.f,0.f,0.f,0.f};
  v8f acc1 = {0.f,0.f,0.f,0.f,0.f,0.f,0.f,0.f};

  for (int k0 = 0; k0 < K; k0 += 32) {
    const float* As; const float* Bs; int kk, lka;
    if (k0 < K1) { As = A1; Bs = B1; kk = k0;       lka = K1; }
    else         { As = A2; Bs = B2; kk = k0 - K1;  lka = K2; }

    { // stage A tile (64x32), fp32 -> bf16
      const float* src = As + (size_t)(bm + arow) * lka + kk + acg;
      #pragma unroll
      for (int j = 0; j < 8; ++j) lsA[arow][acg + j] = (bf16)src[j];
    }
    if (BTRANS) { // B source is [N][K]: contiguous k segments per n-row
      const float* src = Bs + (size_t)(bn + arow) * lka + kk + acg;
      #pragma unroll
      for (int j = 0; j < 8; ++j) lsB[arow][acg + j] = (bf16)src[j];
    } else {      // B source is [K][N]: transpose into LDS
      const float* src = Bs + (size_t)(kk + bkr) * N + bn + bng;
      #pragma unroll
      for (int j = 0; j < 8; ++j) lsB[bng + j][bkr] = (bf16)src[j];
    }
    __syncthreads();

    v16bf af  = load_fragA(&lsA[wm * 16 + ln][0], h);
    v16bf bf0 = load_fragB(&lsB[wnb * 16 + ln][0], h);
    v16bf bf1 = load_fragB(&lsB[(wnb + 1) * 16 + ln][0], h);
    acc0 = __builtin_amdgcn_wmma_f32_16x16x32_bf16(false, af, false, bf0,
                                                   (short)0, acc0, false, false);
    acc1 = __builtin_amdgcn_wmma_f32_16x16x32_bf16(false, af, false, bf1,
                                                   (short)0, acc1, false, false);
    __syncthreads();
  }

  #pragma unroll
  for (int r = 0; r < 8; ++r) {
    const int mg = bm + wm * 16 + h * 8 + r;   // C layout: M = r + 8*half
    const int n0 = bn + wnb * 16 + ln;
    const int n1 = n0 + 16;
    float v0 = acc0[r], v1 = acc1[r];
    if (bias1)  { v0 += bias1[n0];  v1 += bias1[n1]; }
    if (bias2)  { v0 += bias2[n0];  v1 += bias2[n1]; }
    if (NSCALE) { v0 *= nscale[n0]; v1 *= nscale[n1]; }
    if (RELU)   { v0 = fmaxf(v0, 0.f); v1 = fmaxf(v1, 0.f); }
    C[(size_t)mg * N + n0] = v0;
    C[(size_t)mg * N + n1] = v1;
  }
}

// ---------------------------------------------------------------------------
// Elementwise / reduction kernels
// ---------------------------------------------------------------------------

// Per-row 1/max(||mem_keys[r]||,eps) + fused copy into new_keys.
__global__ __launch_bounds__(256)
void key_norm_copy(const float* __restrict__ keys, float* __restrict__ newk,
                   float* __restrict__ kinv)
{
  __shared__ float red[256];
  const int r = blockIdx.x, t = threadIdx.x;
  float ss = 0.f;
  for (int j = t; j < KEY_DIM; j += 256) {
    float v = keys[(size_t)r * KEY_DIM + j];
    newk[(size_t)r * KEY_DIM + j] = v;
    ss += v * v;
  }
  red[t] = ss; __syncthreads();
  for (int s = 128; s > 0; s >>= 1) { if (t < s) red[t] += red[t + s]; __syncthreads(); }
  if (t == 0) kinv[r] = 1.f / fmaxf(sqrtf(red[0]), EPSV);
}

// qn = barcode / max(||barcode||, eps)
__global__ __launch_bounds__(256)
void q_norm(const float* __restrict__ q, float* __restrict__ qn)
{
  __shared__ float red[256];
  const int b = blockIdx.x, t = threadIdx.x;
  float ss = 0.f;
  for (int j = t; j < KEY_DIM; j += 256) { float v = q[b * KEY_DIM + j]; ss += v * v; }
  red[t] = ss; __syncthreads();
  for (int s = 128; s > 0; s >>= 1) { if (t < s) red[t] += red[t + s]; __syncthreads(); }
  const float inv = 1.f / fmaxf(sqrtf(red[0]), EPSV);
  for (int j = t; j < KEY_DIM; j += 256) qn[b * KEY_DIM + j] = q[b * KEY_DIM + j] * inv;
}

// Row argmax over sims (first-index tie-break)
__global__ __launch_bounds__(256)
void row_argmax(const float* __restrict__ sims, int* __restrict__ best,
                float* __restrict__ best_f)
{
  __shared__ float sv[256]; __shared__ int si[256];
  const int b = blockIdx.x, t = threadIdx.x;
  float bv = -3.402823466e38f; int bi = 0x7fffffff;
  for (int n = t; n < DICT_LEN; n += 256) {
    float v = sims[(size_t)b * DICT_LEN + n];
    if (v > bv) { bv = v; bi = n; }
  }
  sv[t] = bv; si[t] = bi; __syncthreads();
  for (int s = 128; s > 0; s >>= 1) {
    if (t < s) {
      if (sv[t + s] > sv[t] || (sv[t + s] == sv[t] && si[t + s] < si[t])) {
        sv[t] = sv[t + s]; si[t] = si[t + s];
      }
    }
    __syncthreads();
  }
  if (t == 0) { best[b] = si[0]; best_f[b] = (float)si[0]; }
}

__device__ inline float sigm(float x) { return 1.f / (1.f + expf(-x)); }

// LSTM cell with DND retrieval
__global__ __launch_bounds__(256)
void lstm_cell(const float* __restrict__ preact, const float* __restrict__ cold,
               const float* __restrict__ mem_vals, const int* __restrict__ best,
               float* __restrict__ out_h, float* __restrict__ out_c)
{
  const int b = blockIdx.x, t = threadIdx.x;
  const size_t base = (size_t)b * (5 * HSZ);
  const size_t mrow = (size_t)best[b] * HSZ;
  for (int j = t; j < HSZ; j += 256) {
    float f  = sigm(preact[base + j]);
    float ig = sigm(preact[base + HSZ + j]);
    float o  = sigm(preact[base + 2 * HSZ + j]);
    float r  = sigm(preact[base + 3 * HSZ + j]);
    float cn = tanhf(preact[base + 4 * HSZ + j]);
    float m  = tanhf(mem_vals[mrow + j]);
    float ct = f * cold[b * HSZ + j] + ig * cn + r * m;
    out_c[b * HSZ + j] = ct;
    out_h[b * HSZ + j] = o * tanhf(ct);
  }
}

// Bulk copy (float4 grid-stride): mem_vals -> new_vals
__global__ __launch_bounds__(256)
void copy_f4(const float4* __restrict__ src, float4* __restrict__ dst, int n4)
{
  int i = blockIdx.x * 256 + threadIdx.x;
  int stride = gridDim.x * 256;
  for (; i < n4; i += stride) dst[i] = src[i];
}

// Serialized scatter: deterministic "last b wins" for duplicate write_idx.
__global__ __launch_bounds__(256)
void scatter_rows(const int* __restrict__ widx, const float* __restrict__ barcode,
                  const float* __restrict__ ct,
                  float* __restrict__ newk, float* __restrict__ newv)
{
  const int t = threadIdx.x;
  for (int b = 0; b < BATCH; ++b) {
    const size_t row = (size_t)widx[b];
    for (int j = t; j < KEY_DIM; j += 256) newk[row * KEY_DIM + j] = barcode[b * KEY_DIM + j];
    for (int j = t; j < HSZ; j += 256)     newv[row * HSZ + j]     = ct[b * HSZ + j];
  }
}

// v_t = ha @ W_critic + b_critic
__global__ __launch_bounds__(256)
void critic(const float* __restrict__ ha, const float* __restrict__ Wc,
            const float* __restrict__ bc, float* __restrict__ out_v)
{
  __shared__ float red[256];
  const int b = blockIdx.x, t = threadIdx.x;
  float s = ha[b * A2C_H + t] * Wc[t] + ha[b * A2C_H + 256 + t] * Wc[256 + t];
  red[t] = s; __syncthreads();
  for (int st = 128; st > 0; st >>= 1) { if (t < st) red[t] += red[t + st]; __syncthreads(); }
  if (t == 0) out_v[b] = red[0] + bc[0];
}

// ---------------------------------------------------------------------------
// Threefry2x32 (JAX PRNG) for jax.random.categorical(key(1), log_pi)
// key(1) -> (k0,k1) = (0,1). counts = iota(B*N_ACT), split at half=2^17.
// ---------------------------------------------------------------------------
__device__ inline unsigned rotl32(unsigned x, int n) { return (x << n) | (x >> (32 - n)); }

__device__ inline unsigned threefry_bits(unsigned flat)
{
  const unsigned k0 = 0u, k1 = 1u;
  const unsigned ks2 = 0x1BD11BDAu ^ k0 ^ k1;
  unsigned p = flat & 131071u;          // pair index (half = 2^17)
  unsigned sel = flat >> 17;            // 0 -> x0 stream, 1 -> x1 stream
  unsigned x0 = p, x1 = p + 131072u;
  x0 += k0; x1 += k1;
#define TFR(r) { x0 += x1; x1 = rotl32(x1, r); x1 ^= x0; }
  TFR(13) TFR(15) TFR(26) TFR(6)   x0 += k1;  x1 += ks2 + 1u;
  TFR(17) TFR(29) TFR(16) TFR(24)  x0 += ks2; x1 += k0 + 2u;
  TFR(13) TFR(15) TFR(26) TFR(6)   x0 += k0;  x1 += k1 + 3u;
  TFR(17) TFR(29) TFR(16) TFR(24)  x0 += k1;  x1 += ks2 + 4u;
  TFR(13) TFR(15) TFR(26) TFR(6)   x0 += ks2; x1 += k0 + 5u;
#undef TFR
  return sel ? x1 : x0;
}

__device__ inline float gumbel_sample(unsigned flat)
{
  const float TINY = 1.17549435e-38f;
  unsigned bits = threefry_bits(flat);
  float f;
  unsigned m = (bits >> 9) | 0x3f800000u;
  __builtin_memcpy(&f, &m, 4);
  float u = (f - 1.0f) * (1.0f - TINY) + TINY;   // uniform in [tiny, 1)
  return -logf(-logf(u));
}

// log_softmax + entropy + Gumbel-max sampling + prob_a_t
__global__ __launch_bounds__(256)
void softmax_sample(const float* __restrict__ logits,
                    float* __restrict__ out_a, float* __restrict__ out_prob,
                    float* __restrict__ out_ent)
{
  __shared__ float red[256]; __shared__ int ridx[256];
  const int b = blockIdx.x, t = threadIdx.x;
  const float* lrow = logits + (size_t)b * N_ACT;

  // row max
  float mx = -3.402823466e38f;
  for (int j = t; j < N_ACT; j += 256) mx = fmaxf(mx, lrow[j]);
  red[t] = mx; __syncthreads();
  for (int s = 128; s > 0; s >>= 1) { if (t < s) red[t] = fmaxf(red[t], red[t + s]); __syncthreads(); }
  const float rowmax = red[0]; __syncthreads();

  // log-sum-exp
  float se = 0.f;
  for (int j = t; j < N_ACT; j += 256) se += expf(lrow[j] - rowmax);
  red[t] = se; __syncthreads();
  for (int s = 128; s > 0; s >>= 1) { if (t < s) red[t] += red[t + s]; __syncthreads(); }
  const float logZ = rowmax + logf(red[0]); __syncthreads();

  // entropy = -sum pi * log_pi
  float es = 0.f;
  for (int j = t; j < N_ACT; j += 256) { float lp = lrow[j] - logZ; es += expf(lp) * lp; }
  red[t] = es; __syncthreads();
  for (int s = 128; s > 0; s >>= 1) { if (t < s) red[t] += red[t + s]; __syncthreads(); }
  const float entropy = -red[0]; __syncthreads();

  // Gumbel-max argmax (first-index tie-break)
  float bv = -3.402823466e38f; int bi = 0x7fffffff;
  for (int j = t; j < N_ACT; j += 256) {
    float g  = gumbel_sample((unsigned)(b * N_ACT + j));
    float sc = (lrow[j] - logZ) + g;
    if (sc > bv) { bv = sc; bi = j; }
  }
  red[t] = bv; ridx[t] = bi; __syncthreads();
  for (int s = 128; s > 0; s >>= 1) {
    if (t < s) {
      if (red[t + s] > red[t] || (red[t + s] == red[t] && ridx[t + s] < ridx[t])) {
        red[t] = red[t + s]; ridx[t] = ridx[t + s];
      }
    }
    __syncthreads();
  }
  if (t == 0) {
    int a = ridx[0];
    out_a[b]    = (float)a;
    out_prob[b] = lrow[a] - logZ;
    out_ent[b]  = entropy;
  }
}

// ---------------------------------------------------------------------------
// Launch
// ---------------------------------------------------------------------------
extern "C" void kernel_launch(void* const* d_in, const int* in_sizes, int n_in,
                              void* d_out, int out_size, void* d_ws, size_t ws_size,
                              hipStream_t stream)
{
  const float* obs      = (const float*)d_in[0];
  const float* barcode  = (const float*)d_in[1];
  const float* h_in     = (const float*)d_in[2];
  const float* c_in     = (const float*)d_in[3];
  const int*   widx     = (const int*)  d_in[4];
  const float* Wi       = (const float*)d_in[5];
  const float* bi       = (const float*)d_in[6];
  const float* Wh       = (const float*)d_in[7];
  const float* bh       = (const float*)d_in[8];
  const float* mem_keys = (const float*)d_in[9];
  const float* mem_vals = (const float*)d_in[10];
  const float* Wa       = (const float*)d_in[11];
  const float* ba       = (const float*)d_in[12];
  const float* W_actor  = (const float*)d_in[13];
  const float* b_actor  = (const float*)d_in[14];
  const float* W_critic = (const float*)d_in[15];
  const float* b_critic = (const float*)d_in[16];

  float* out = (float*)d_out;
  float* ws  = (float*)d_ws;

  float* preact = ws + W_PREACT;
  float* sims   = ws + W_SIMS;
  float* qn     = ws + W_QN;
  float* kinv   = ws + W_KINV;
  float* ha     = ws + W_HA;
  float* logits = ws + W_LOGITS;
  int*   best   = (int*)(ws + W_BEST);

  // 1) key norms + new_keys copy; mem_vals copy
  key_norm_copy<<<DICT_LEN, 256, 0, stream>>>(mem_keys, out + O_NEWK, kinv);
  copy_f4<<<2048, 256, 0, stream>>>((const float4*)mem_vals, (float4*)(out + O_NEWV),
                                    DICT_LEN * HSZ / 4);
  // 2) query normalization
  q_norm<<<BATCH, 256, 0, stream>>>(barcode, qn);

  // 3) sims = qn @ mem_keys^T, scaled per-column by 1/||key|| (WMMA, BTRANS)
  gemm_wmma_bf16<true, false, true><<<dim3(DICT_LEN / 64, BATCH / 64), 256, 0, stream>>>(
      qn, nullptr, KEY_DIM, 0, mem_keys, nullptr, nullptr, nullptr, kinv,
      sims, DICT_LEN);

  // 4) argmax over dictionary
  row_argmax<<<BATCH, 256, 0, stream>>>(sims, best, out + O_BEST);

  // 5) LSTM preact = [obs|h] @ [Wi;Wh] + bi + bh (WMMA, two K segments)
  gemm_wmma_bf16<false, false, false><<<dim3(5 * HSZ / 64, BATCH / 64), 256, 0, stream>>>(
      obs, h_in, DIN, HSZ, Wi, Wh, bi, bh, nullptr, preact, 5 * HSZ);

  // 6) cell update -> h_t, c_t
  lstm_cell<<<BATCH, 256, 0, stream>>>(preact, c_in, mem_vals, best,
                                       out + O_H, out + O_C);

  // 7) deterministic scatter of barcode / c_t into new_keys / new_vals
  scatter_rows<<<1, 256, 0, stream>>>(widx, barcode, out + O_C,
                                      out + O_NEWK, out + O_NEWV);

  // 8) ha = relu(c_t @ Wa + ba)
  gemm_wmma_bf16<false, true, false><<<dim3(A2C_H / 64, BATCH / 64), 256, 0, stream>>>(
      out + O_C, nullptr, HSZ, 0, Wa, nullptr, ba, nullptr, nullptr, ha, A2C_H);

  // 9) logits = ha @ W_actor + b_actor
  gemm_wmma_bf16<false, false, false><<<dim3(N_ACT / 64, BATCH / 64), 256, 0, stream>>>(
      ha, nullptr, A2C_H, 0, W_actor, nullptr, b_actor, nullptr, nullptr,
      logits, N_ACT);

  // 10) critic head
  critic<<<BATCH, 256, 0, stream>>>(ha, W_critic, b_critic, out + O_V);

  // 11) log_softmax + entropy + threefry/Gumbel sampling
  softmax_sample<<<BATCH, 256, 0, stream>>>(logits, out + O_A, out + O_PROB,
                                            out + O_ENT);
}
